// MultiHeadAttention_45234595562045
// MI455X (gfx1250) — matrix operand order repaired and checked by reading, not compiled, not run on hardware
//
#include <hip/hip_runtime.h>
#include <hip/hip_bf16.h>

// Fixed shapes from the reference.
constexpr int B_ = 4, T_ = 2048, D_ = 1024, H_ = 16, HS_ = 64;

typedef __attribute__((ext_vector_type(16))) __bf16       v16bf;
typedef __attribute__((ext_vector_type(8)))  float        v8f;
typedef __attribute__((ext_vector_type(4)))  unsigned int u32x4;
typedef __attribute__((ext_vector_type(2)))  unsigned int u32x2;
typedef __attribute__((ext_vector_type(4)))  float        f32x4;
typedef __attribute__((ext_vector_type(4)))  __bf16       bf16x4;

__device__ __forceinline__ __bf16 f2bf(float f) { return (__bf16)f; }

// v_perm_b32: interleave the low / high bf16 halves of two dwords.
// result low half = a's half, high half = b's half (memory order k, k+1).
__device__ __forceinline__ unsigned int ilv_lo(unsigned int a, unsigned int b) {
  return __builtin_amdgcn_perm(b, a, 0x05040100u);
}
__device__ __forceinline__ unsigned int ilv_hi(unsigned int a, unsigned int b) {
  return __builtin_amdgcn_perm(b, a, 0x07060302u);
}

// A fragment: 16x32 (MxK) bf16 tile, row-major, row stride ld (multiple of 8).
// Lane's 16 values = two contiguous 8-element runs of row m=lane&15:
// K = c0..c0+7 and 16+c0..+7, c0 = 8*(lane>=16)  ->  2x ds_load_b128.
__device__ __forceinline__ v16bf lds_load_Afrag(const __bf16* p, int ld, int lane) {
  const __bf16* q = p + (lane & 15) * ld + (((lane >> 4) & 1) << 3);
  union { u32x4 q4[2]; v16bf v; } r;
  r.q4[0] = *(const u32x4*)(q);
  r.q4[1] = *(const u32x4*)(q + 16);
  return r.v;
}

// B fragment from K-contiguous transposed tile BT[n][k] (32 k per row):
// lane holds K = c0..c0+7 and 16+c0..+7 of column n=lane&15, c0 = 8*(lane>=16).
__device__ __forceinline__ v16bf lds_load_Bfrag(const __bf16* p, int ld, int lane) {
  const __bf16* q = p + (lane & 15) * ld + (((lane >> 4) & 1) << 3);
  union { u32x4 q4[2]; v16bf v; } r;
  r.q4[0] = *(const u32x4*)(q);
  r.q4[1] = *(const u32x4*)(q + 16);
  return r.v;
}

__device__ __forceinline__ v8f wmma_bf16(v16bf a, v16bf b, v8f c) {
  return __builtin_amdgcn_wmma_f32_16x16x32_bf16(false, a, false, b, (short)0, c,
                                                 false, false);
}

// ---------------------------------------------------------------------------
// Kernel 0: bulk f32 -> bf16 (packed v_cvt_pk_bf16_f32 via convertvector).
// ---------------------------------------------------------------------------
__global__ __launch_bounds__(256) void cvt_f32_bf16_kernel(
    const float* __restrict__ src, __bf16* __restrict__ dst, unsigned int n4) {
  const unsigned int i = blockIdx.x * 256u + threadIdx.x;
  if (i >= n4) return;
  f32x4 f = *(const f32x4*)(src + (size_t)i * 4);
  bf16x4 b = __builtin_convertvector(f, bf16x4);
  *(u32x2*)(dst + (size_t)i * 4) = __builtin_bit_cast(u32x2, b);
}

// ---------------------------------------------------------------------------
// Kernel 1: QKV projection.  grid = (T/128, H, 3*B), block = 256 (8 waves).
// All-bf16 inputs, double-buffered LDS, one barrier per K step.
// ---------------------------------------------------------------------------
__global__ __launch_bounds__(256) void qkv_proj_kernel(
    const __bf16* __restrict__ xbf, const __bf16* __restrict__ wqkv,
    __bf16* __restrict__ qkv) {
  constexpr int LDA = 40;                        // 80B rows (16B aligned)
  __shared__ __align__(16) __bf16 As[2][128 * LDA];
  __shared__ __align__(16) __bf16 BT[2][64 * LDA];   // BT[n][k]

  const int tid  = threadIdx.x;
  const int lane = tid & 31;
  const int wave = tid >> 5;
  const int mw   = wave & 3;
  const int nw   = wave >> 2;
  const int tt   = blockIdx.x;
  const int h    = blockIdx.y;
  const int b    = blockIdx.z % B_;
  const int mat  = blockIdx.z / B_;

  const __bf16* xrow = xbf + ((size_t)b * T_ + (size_t)tt * 128) * D_;
  const __bf16* wrow = wqkv + ((size_t)mat * H_ + h) * D_ * HS_;

  // per-thread tile slots
  const int rowA = tid >> 2, c8A = (tid & 3) << 3;   // A: 2 slots (rowA, rowA+64)
  const int kpW  = tid >> 4, c4W = (tid & 15) << 2;  // W: 1 k-pair x 4 n

  u32x4 va0, va1; u32x2 w0, w1;
  auto load_ab = [&](int kk) {
    const __bf16* ga = xrow + (size_t)rowA * D_ + kk + c8A;
    __builtin_prefetch(ga + 64, 0, 1);               // speculative next-next slice
    va0 = *(const u32x4*)(ga);
    va1 = *(const u32x4*)(ga + (size_t)64 * D_);
    const __bf16* g0 = wrow + (size_t)(kk + 2 * kpW) * HS_ + c4W;
    w0 = *(const u32x2*)(g0);
    w1 = *(const u32x2*)(g0 + HS_);
  };
  auto store_ab = [&](int p) {
    *(u32x4*)(As[p] + rowA * LDA + c8A) = va0;
    *(u32x4*)(As[p] + (rowA + 64) * LDA + c8A) = va1;
    *(unsigned int*)(BT[p] + (c4W + 0) * LDA + 2 * kpW) = ilv_lo(w0[0], w1[0]);
    *(unsigned int*)(BT[p] + (c4W + 1) * LDA + 2 * kpW) = ilv_hi(w0[0], w1[0]);
    *(unsigned int*)(BT[p] + (c4W + 2) * LDA + 2 * kpW) = ilv_lo(w0[1], w1[1]);
    *(unsigned int*)(BT[p] + (c4W + 3) * LDA + 2 * kpW) = ilv_hi(w0[1], w1[1]);
  };

  v8f acc[2][2] = {};
  load_ab(0);
  int p = 0;
#pragma unroll 2
  for (int k0 = 0; k0 < D_; k0 += 32) {
    store_ab(p);
    __syncthreads();
    if (k0 + 32 < D_) load_ab(k0 + 32);            // overlaps with compute
    v16bf a0 = lds_load_Afrag(As[p] + (mw * 32) * LDA, LDA, lane);
    v16bf a1 = lds_load_Afrag(As[p] + (mw * 32 + 16) * LDA, LDA, lane);
    v16bf b0 = lds_load_Bfrag(BT[p] + (nw * 32) * LDA, LDA, lane);
    v16bf b1 = lds_load_Bfrag(BT[p] + (nw * 32 + 16) * LDA, LDA, lane);
    acc[0][0] = wmma_bf16(a0, b0, acc[0][0]);
    acc[0][1] = wmma_bf16(a0, b1, acc[0][1]);
    acc[1][0] = wmma_bf16(a1, b0, acc[1][0]);
    acc[1][1] = wmma_bf16(a1, b1, acc[1][1]);
    p ^= 1;
  }

  __bf16* outp = qkv + (size_t)mat * B_ * H_ * T_ * HS_ +
                 ((size_t)(b * H_ + h) * T_ + (size_t)tt * 128) * HS_;
  const int mofs = (lane >> 4) << 3;
  const int nn   = lane & 15;
#pragma unroll
  for (int i = 0; i < 2; ++i)
#pragma unroll
    for (int j = 0; j < 2; ++j)
#pragma unroll
      for (int r = 0; r < 8; ++r) {
        const int m = mw * 32 + i * 16 + r + mofs;
        const int n = nw * 32 + j * 16 + nn;
        outp[(size_t)m * HS_ + n] = f2bf(acc[i][j][r]);
      }
}

// ---------------------------------------------------------------------------
// Kernel 2: causal flash attention.  grid = (T/64, H, B), block = 128 (4 waves).
// ---------------------------------------------------------------------------
__global__ __launch_bounds__(128) void attn_kernel(
    const __bf16* __restrict__ qkv, __bf16* __restrict__ attn) {
  constexpr int LD = 72;                         // 144B rows (16B aligned)
  __shared__ __align__(16) __bf16 Qs[64 * LD];   // Q[m][e]
  __shared__ __align__(16) __bf16 Ks[64 * LD];   // K[kn][e]  (B^T for S)
  __shared__ __align__(16) __bf16 VT[64 * LD];   // V^T[e][kn] (B^T for O)
  __shared__ __align__(16) __bf16 Ps[64 * LD];   // P[m][kn], per-wave strips

  const int tid  = threadIdx.x;
  const int lane = tid & 31;
  const int wave = tid >> 5;
  const int qt   = blockIdx.x;
  const int h    = blockIdx.y;
  const int b    = blockIdx.z;

  const size_t head = ((size_t)(b * H_ + h)) * T_ * HS_;
  const size_t msz  = (size_t)B_ * H_ * T_ * HS_;
  const __bf16* Qg = qkv + head;
  const __bf16* Kg = qkv + msz + head;
  const __bf16* Vg = qkv + 2 * msz + head;
  const int q_base = qt * 64;

  // Q tile 64x64 bf16 (512 x 16B slots, 4/thread)
#pragma unroll
  for (int i = 0; i < 4; ++i) {
    const int idx = tid + i * 128;
    const int row = idx >> 3;
    const int c8  = (idx & 7) << 3;
    *(u32x4*)(Qs + row * LD + c8) =
        *(const u32x4*)(Qg + (size_t)(q_base + row) * HS_ + c8);
  }
  __syncthreads();
  const v16bf aq0 = lds_load_Afrag(Qs + wave * 16 * LD, LD, lane);
  const v16bf aq1 = lds_load_Afrag(Qs + wave * 16 * LD + 32, LD, lane);

  float mstate[8], lstate[8];
  v8f o[4] = {};
#pragma unroll
  for (int r = 0; r < 8; ++r) { mstate[r] = -__builtin_inff(); lstate[r] = 0.f; }

  const int mofs = (lane >> 4) << 3;
  const int nn   = lane & 15;

  for (int j = 0; j <= qt; ++j) {
    const int kv = j * 64;
    // phase 1: issue all global loads (overlap previous compute tail)
    u32x4 kreg[4];
    u32x2 vr0[4], vr1[4];
#pragma unroll
    for (int i = 0; i < 4; ++i) {
      const int idx = tid + i * 128;
      const int row = idx >> 3;
      const int c8  = (idx & 7) << 3;
      kreg[i] = *(const u32x4*)(Kg + (size_t)(kv + row) * HS_ + c8);
    }
#pragma unroll
    for (int i = 0; i < 4; ++i) {
      const int idx = tid + i * 128;
      const int kp  = idx >> 4;
      const int c4  = (idx & 15) << 2;
      vr0[i] = *(const u32x2*)(Vg + (size_t)(kv + 2 * kp) * HS_ + c4);
      vr1[i] = *(const u32x2*)(Vg + (size_t)(kv + 2 * kp + 1) * HS_ + c4);
    }
    __syncthreads();                 // previous tile consumers done
    // phase 2: LDS stores
#pragma unroll
    for (int i = 0; i < 4; ++i) {
      const int idx = tid + i * 128;
      const int row = idx >> 3;
      const int c8  = (idx & 7) << 3;
      *(u32x4*)(Ks + row * LD + c8) = kreg[i];
    }
#pragma unroll
    for (int i = 0; i < 4; ++i) {
      const int idx = tid + i * 128;
      const int kp  = idx >> 4;
      const int c4  = (idx & 15) << 2;
      *(unsigned int*)(VT + (c4 + 0) * LD + 2 * kp) = ilv_lo(vr0[i][0], vr1[i][0]);
      *(unsigned int*)(VT + (c4 + 1) * LD + 2 * kp) = ilv_hi(vr0[i][0], vr1[i][0]);
      *(unsigned int*)(VT + (c4 + 2) * LD + 2 * kp) = ilv_lo(vr0[i][1], vr1[i][1]);
      *(unsigned int*)(VT + (c4 + 3) * LD + 2 * kp) = ilv_hi(vr0[i][1], vr1[i][1]);
    }
    __syncthreads();

    // S = Q * K^T
    v8f s[4];
#pragma unroll
    for (int tn = 0; tn < 4; ++tn) {
      v8f sa = {};
      sa = wmma_bf16(aq0, lds_load_Bfrag(Ks + tn * 16 * LD, LD, lane), sa);
      sa = wmma_bf16(aq1, lds_load_Bfrag(Ks + tn * 16 * LD + 32, LD, lane), sa);
      s[tn] = sa;
    }

    // online softmax
    const int rowg0 = q_base + wave * 16 + mofs;
    const int colg  = kv + nn;
#pragma unroll
    for (int r = 0; r < 8; ++r) {
      const int rowg = rowg0 + r;
      float rm = -__builtin_inff();
#pragma unroll
      for (int tn = 0; tn < 4; ++tn) {
        float sv = s[tn][r] * 0.125f;                  // 1/sqrt(HS)
        sv = (colg + tn * 16 <= rowg) ? sv : -__builtin_inff();
        s[tn][r] = sv;
        rm = fmaxf(rm, sv);
      }
      rm = fmaxf(rm, __shfl_xor(rm, 1, 32));
      rm = fmaxf(rm, __shfl_xor(rm, 2, 32));
      rm = fmaxf(rm, __shfl_xor(rm, 4, 32));
      rm = fmaxf(rm, __shfl_xor(rm, 8, 32));
      const float mnew  = fmaxf(mstate[r], rm);
      const float alpha = __expf(mstate[r] - mnew);
      float psum = 0.f;
#pragma unroll
      for (int tn = 0; tn < 4; ++tn) {
        const float p = __expf(s[tn][r] - mnew);
        s[tn][r] = p;
        psum += p;
      }
      psum += __shfl_xor(psum, 1, 32);
      psum += __shfl_xor(psum, 2, 32);
      psum += __shfl_xor(psum, 4, 32);
      psum += __shfl_xor(psum, 8, 32);
      lstate[r] = lstate[r] * alpha + psum;
      mstate[r] = mnew;
#pragma unroll
      for (int te = 0; te < 4; ++te) o[te][r] *= alpha;
    }

    // stage P (C-layout -> A-layout) into this wave's private strip
    const int pm = wave * 16 + mofs;
#pragma unroll
    for (int r = 0; r < 8; ++r)
#pragma unroll
      for (int tn = 0; tn < 4; ++tn)
        Ps[(pm + r) * LD + tn * 16 + nn] = f2bf(s[tn][r]);

    // O += P * V
    v16bf ap0 = lds_load_Afrag(Ps + wave * 16 * LD, LD, lane);
    v16bf ap1 = lds_load_Afrag(Ps + wave * 16 * LD + 32, LD, lane);
#pragma unroll
    for (int te = 0; te < 4; ++te) {
      o[te] = wmma_bf16(ap0, lds_load_Bfrag(VT + te * 16 * LD, LD, lane), o[te]);
      o[te] = wmma_bf16(ap1, lds_load_Bfrag(VT + te * 16 * LD + 32, LD, lane), o[te]);
    }
  }

  // normalize and store head-concatenated: attn[b][t][h*64+e]
  __bf16* orow =
      attn + ((size_t)b * T_ + q_base + wave * 16) * D_ + (size_t)h * HS_;
#pragma unroll
  for (int r = 0; r < 8; ++r) {
    const int m = r + mofs;
    const float inv_l = 1.0f / lstate[r];
#pragma unroll
    for (int te = 0; te < 4; ++te)
      orow[(size_t)m * D_ + te * 16 + nn] = f2bf(o[te][r] * inv_l);
  }
}

// ---------------------------------------------------------------------------
// Kernel 3: output projection + bias.  grid = (B*T/128, D/64), block = 256.
// Wp (pre-converted bf16) is [n][d]: already the B^T layout -> pure copies.
// ---------------------------------------------------------------------------
__global__ __launch_bounds__(256) void out_proj_kernel(
    const __bf16* __restrict__ attn, const __bf16* __restrict__ wp,
    const float* __restrict__ bp, float* __restrict__ out) {
  constexpr int LDA = 40;
  __shared__ __align__(16) __bf16 As[2][128 * LDA];
  __shared__ __align__(16) __bf16 BT[2][64 * LDA];

  const int tid  = threadIdx.x;
  const int lane = tid & 31;
  const int wave = tid >> 5;
  const int mw   = wave & 3;
  const int nw   = wave >> 2;
  const int mbase = blockIdx.x * 128;
  const int nbase = blockIdx.y * 64;

  const int rowA = tid >> 2, c8A = (tid & 3) << 3;   // A: 2 slots
  const int nB   = tid >> 2, c8B = (tid & 3) << 3;   // B: 1 slot

  u32x4 va0, va1, vb;
  auto load_ab = [&](int kk) {
    const __bf16* ga = attn + (size_t)(mbase + rowA) * D_ + kk + c8A;
    __builtin_prefetch(ga + 64, 0, 1);
    va0 = *(const u32x4*)(ga);
    va1 = *(const u32x4*)(ga + (size_t)64 * D_);
    vb  = *(const u32x4*)(wp + (size_t)(nbase + nB) * D_ + kk + c8B);
  };
  auto store_ab = [&](int p) {
    *(u32x4*)(As[p] + rowA * LDA + c8A) = va0;
    *(u32x4*)(As[p] + (rowA + 64) * LDA + c8A) = va1;
    *(u32x4*)(BT[p] + nB * LDA + c8B) = vb;
  };

  v8f acc[2][2] = {};
  load_ab(0);
  int p = 0;
#pragma unroll 2
  for (int k0 = 0; k0 < D_; k0 += 32) {
    store_ab(p);
    __syncthreads();
    if (k0 + 32 < D_) load_ab(k0 + 32);
    v16bf a0 = lds_load_Afrag(As[p] + (mw * 32) * LDA, LDA, lane);
    v16bf a1 = lds_load_Afrag(As[p] + (mw * 32 + 16) * LDA, LDA, lane);
    v16bf b0 = lds_load_Bfrag(BT[p] + (nw * 32) * LDA, LDA, lane);
    v16bf b1 = lds_load_Bfrag(BT[p] + (nw * 32 + 16) * LDA, LDA, lane);
    acc[0][0] = wmma_bf16(a0, b0, acc[0][0]);
    acc[0][1] = wmma_bf16(a0, b1, acc[0][1]);
    acc[1][0] = wmma_bf16(a1, b0, acc[1][0]);
    acc[1][1] = wmma_bf16(a1, b1, acc[1][1]);
    p ^= 1;
  }

  const int mofs = (lane >> 4) << 3;
  const int nn   = lane & 15;
#pragma unroll
  for (int i = 0; i < 2; ++i)
#pragma unroll
    for (int j = 0; j < 2; ++j)
#pragma unroll
      for (int r = 0; r < 8; ++r) {
        const int m = mw * 32 + i * 16 + r + mofs;
        const int c = nw * 32 + j * 16 + nn;
        out[(size_t)(mbase + m) * D_ + nbase + c] = acc[i][j][r] + bp[nbase + c];
      }
}

// ---------------------------------------------------------------------------
extern "C" void kernel_launch(void* const* d_in, const int* in_sizes, int n_in,
                              void* d_out, int out_size, void* d_ws, size_t ws_size,
                              hipStream_t stream) {
  (void)in_sizes; (void)n_in; (void)out_size; (void)ws_size;
  const float* x  = (const float*)d_in[0];
  const float* Wq = (const float*)d_in[1];
  const float* Wk = (const float*)d_in[2];
  const float* Wv = (const float*)d_in[3];
  const float* Wp = (const float*)d_in[4];
  const float* bp = (const float*)d_in[5];
  float* out = (float*)d_out;

  // workspace layout (bf16 elements), ~92 MB total
  const size_t XN = (size_t)B_ * T_ * D_;        // 8,388,608
  const size_t WN = (size_t)H_ * D_ * HS_;       // 1,048,576 (per mat)
  const size_t PN = (size_t)D_ * D_;             // 1,048,576
  const size_t MN = (size_t)B_ * H_ * T_ * HS_;  // 8,388,608 (per mat)
  __bf16* xbf  = (__bf16*)d_ws;
  __bf16* wqkv = xbf + XN;
  __bf16* wpbf = wqkv + 3 * WN;
  __bf16* qkv  = wpbf + PN;
  __bf16* attn = qkv + 3 * MN;

  // pre-convert inputs to bf16 once
  cvt_f32_bf16_kernel<<<dim3((unsigned)(XN / 4 / 256)), dim3(256), 0, stream>>>(
      x, xbf, (unsigned)(XN / 4));
  cvt_f32_bf16_kernel<<<dim3((unsigned)(WN / 4 / 256)), dim3(256), 0, stream>>>(
      Wq, wqkv + 0 * WN, (unsigned)(WN / 4));
  cvt_f32_bf16_kernel<<<dim3((unsigned)(WN / 4 / 256)), dim3(256), 0, stream>>>(
      Wk, wqkv + 1 * WN, (unsigned)(WN / 4));
  cvt_f32_bf16_kernel<<<dim3((unsigned)(WN / 4 / 256)), dim3(256), 0, stream>>>(
      Wv, wqkv + 2 * WN, (unsigned)(WN / 4));
  cvt_f32_bf16_kernel<<<dim3((unsigned)(PN / 4 / 256)), dim3(256), 0, stream>>>(
      Wp, wpbf, (unsigned)(PN / 4));

  dim3 g1(T_ / 128, H_, 3 * B_);
  qkv_proj_kernel<<<g1, dim3(256), 0, stream>>>(xbf, wqkv, qkv);

  dim3 g2(T_ / 64, H_, B_);
  attn_kernel<<<g2, dim3(128), 0, stream>>>(qkv, attn);

  dim3 g3((B_ * T_) / 128, D_ / 64);
  out_proj_kernel<<<g3, dim3(256), 0, stream>>>(attn, wpbf, bp, out);
}
